// MultiHeadedAttention_76905684402629
// MI455X (gfx1250) — matrix operand order repaired and checked by reading, not compiled, not run on hardware
//
#include <hip/hip_runtime.h>

// ---------------- problem constants ----------------
constexpr int Bc = 2, Tc = 2048, Dc = 2048, Hc = 16, DHc = 128;

typedef __attribute__((ext_vector_type(16))) __bf16    bf16x16;
typedef __attribute__((ext_vector_type(8)))  float     f32x8;
typedef __attribute__((ext_vector_type(4)))  unsigned  u32x4;
typedef __attribute__((ext_vector_type(8)))  int       i32x8;
typedef __attribute__((ext_vector_type(4)))  int       i32x4;

__device__ __forceinline__ __bf16 f2bf(float f) {
  unsigned u = __builtin_bit_cast(unsigned, f);
  u += 0x7fffu + ((u >> 16) & 1u);                 // round-to-nearest-even
  unsigned short h = (unsigned short)(u >> 16);
  return __builtin_bit_cast(__bf16, h);
}

// ---------------- kernel 1: f32 -> bf16 ----------------
__global__ void mha_cvt_bf16(const float* __restrict__ src, __bf16* __restrict__ dst, int n) {
  int i = (blockIdx.x * blockDim.x + threadIdx.x) * 4;
  if (i + 3 < n) {
    float4 v = *(const float4*)(src + i);
    dst[i + 0] = f2bf(v.x);
    dst[i + 1] = f2bf(v.y);
    dst[i + 2] = f2bf(v.z);
    dst[i + 3] = f2bf(v.w);
  }
}

// ---------------- GEMM tiling ----------------
#define BM 64
#define BN 128
#define BK 32
#define LDS_A 40   // padded stride (bf16) -> 80B rows, 16B aligned
#define LDS_B 40

// ---------------- kernel 2: fused QKV projection + RoPE ----------------
__global__ __launch_bounds__(256) void mha_qkv_rope(
    const __bf16* __restrict__ X,
    const __bf16* __restrict__ Wq, const __bf16* __restrict__ Wk, const __bf16* __restrict__ Wv,
    __bf16* __restrict__ Qr, __bf16* __restrict__ Kr, __bf16* __restrict__ Vr) {
  __shared__ __align__(16) __bf16 Xs[BM * LDS_A];
  __shared__ __align__(16) __bf16 Ws[BN * LDS_B];

  const int tid = threadIdx.x, lane = tid & 31, wid = tid >> 5;
  const int lh = lane >> 4, ln = lane & 15;
  const int waveM = wid >> 1, waveN = wid & 1;
  const int mBase = blockIdx.y * BM;
  const int nBase = blockIdx.x * BN;
  const int z = blockIdx.z;
  const __bf16* __restrict__ W = (z == 0) ? Wq : (z == 1) ? Wk : Wv;

  f32x8 acc[4] = {};

  for (int k0 = 0; k0 < Dc; k0 += BK) {
    {  // stage X tile 64x32 (one b128 per thread)
      int row = tid >> 2, c = (tid & 3) * 8;
      *(u32x4*)(Xs + row * LDS_A + c) = *(const u32x4*)(X + (size_t)(mBase + row) * Dc + k0 + c);
    }
#pragma unroll
    for (int i = 0; i < 2; ++i) {  // stage W tile 128x32 (two b128 per thread)
      int c2 = tid + i * 256;
      int row = c2 >> 2, c = (c2 & 3) * 8;
      *(u32x4*)(Ws + row * LDS_B + c) = *(const u32x4*)(W + (size_t)(nBase + row) * Dc + k0 + c);
    }
    __syncthreads();

    if (k0 + BK < Dc) {  // lowers to global_prefetch_b8
      __builtin_prefetch(X + (size_t)(mBase + (tid >> 2)) * Dc + k0 + BK, 0, 1);
      __builtin_prefetch(W + (size_t)(nBase + (tid >> 1)) * Dc + k0 + BK, 0, 1);
    }

    bf16x16 a;
#pragma unroll
    for (int e = 0; e < 16; ++e) {
      int kk = 16 * lh + e;                             // A-fragment k pattern
      a[e] = Xs[(waveM * 16 + ln) * LDS_A + kk];
    }
#pragma unroll
    for (int nf = 0; nf < 4; ++nf) {
      bf16x16 bf;
#pragma unroll
      for (int e = 0; e < 16; ++e) {
        int kk = 16 * lh + e;                           // B-fragment k pattern (contiguous)
        bf[e] = Ws[(waveN * 64 + nf * 16 + ln) * LDS_B + kk];
      }
      acc[nf] = __builtin_amdgcn_wmma_f32_16x16x32_bf16(false, a, false, bf, (short)0,
                                                        acc[nf], false, false);
    }
    __syncthreads();
  }

  // epilogue: RoPE (interleaved pairs) for Q/K, bf16 store to [B,H,T,DH]
  const bool isV = (z == 2);
  __bf16* __restrict__ dst = (z == 0) ? Qr : (z == 1) ? Kr : Vr;
#pragma unroll
  for (int nf = 0; nf < 4; ++nf) {
    int n = nBase + waveN * 64 + nf * 16 + ln;   // feature 0..2047
    int hh = n >> 7, fh = n & 127;
    int pairI = fh >> 1;
    float theta = exp2f(-0.20762050593046627f * (float)pairI);  // 10000^(-2i/128)
#pragma unroll
    for (int r = 0; r < 8; ++r) {
      int mtok = mBase + waveM * 16 + 8 * lh + r;      // C layout: M = r + 8*(lane/16)
      int b = mtok >> 11, t = mtok & (Tc - 1);
      float v = acc[nf][r];
      float res;
      if (!isV) {
        float partner = __shfl_xor(v, 1, 32);          // pair element (n^1 lives in lane^1)
        float sn, cs;
        __sincosf((float)t * theta, &sn, &cs);
        res = (fh & 1) ? (sn * partner + cs * v) : (cs * v - sn * partner);
      } else {
        res = v;
      }
      dst[(((size_t)(b * Hc + hh) * Tc + t) * DHc) + fh] = f2bf(res);
    }
  }
}

// ---------------- kernel 3: causal flash attention (transposed WMMA form) --------
#define QB 128       // queries per block (8 waves x 16)
#define KT 32        // keys per tile
#define LKV 136      // padded K/V LDS row stride (bf16): 128 + 4 DWORDs TDM pad

// Issue one TDM 2D tile load: 32 rows x 128 bf16, row stride DHc elements,
// LDS destination padded by 4 DWORDs after every 64 DWORDs (-> LKV stride).
__device__ __forceinline__ void tdm_load_kv_tile(unsigned lds_byte, unsigned long long ga) {
  u32x4 g0;
  g0.x = 1u;                                            // count=1, user descriptor
  g0.y = lds_byte;                                      // lds_addr
  g0.z = (unsigned)(ga & 0xffffffffull);                // global_addr[31:0]
  g0.w = (unsigned)((ga >> 32) & 0x1ffffffull) | (2u << 30);  // global_addr[56:32] | type=2
  i32x8 g1;
  g1[0] = (int)((1u << 16)        // data_size = 2 bytes
              | (1u << 20)        // pad_enable
              | (5u << 22)        // pad_interval: 64 DWORDs
              | (3u << 25));      // pad_amount: 4 DWORDs (16B) -> row stride 272B
  g1[1] = (int)(((unsigned)DHc & 0xffffu) << 16);       // tensor_dim0 = 128
  g1[2] = (int)(((unsigned)Tc  & 0xffffu) << 16);       // tensor_dim1 = 2048 (lo16)
  g1[3] = (int)(((unsigned)DHc & 0xffffu) << 16);       // tile_dim0 = 128
  g1[4] = KT;                                           // tile_dim1 = 32
  g1[5] = DHc;                                          // tensor_dim0_stride = 128
  g1[6] = 0;
  g1[7] = 0;
  i32x4 g2 = {0, 0, 0, 0};
  i32x4 g3 = {0, 0, 0, 0};
#if defined(__clang_major__) && (__clang_major__ >= 23)
  i32x8 g4 = {0, 0, 0, 0, 0, 0, 0, 0};
  __builtin_amdgcn_tensor_load_to_lds(g0, g1, g2, g3, g4, 0);
#else
  __builtin_amdgcn_tensor_load_to_lds(g0, g1, g2, g3, 0);
#endif
}

__global__ __launch_bounds__(256) void mha_attn(
    const __bf16* __restrict__ Qr, const __bf16* __restrict__ Kr,
    const __bf16* __restrict__ Vr, __bf16* __restrict__ Ob) {
  __shared__ __align__(16) __bf16 Ks[2][KT * LKV];
  __shared__ __align__(16) __bf16 Vs[2][KT * LKV];

  const int tid = threadIdx.x, lane = tid & 31, wid = tid >> 5;
  const int lh = lane >> 4, ln = lane & 15;
  const int bh = blockIdx.y;                   // b*H + h
  const int qbase = blockIdx.x * QB;
  const int q = qbase + wid * 16 + ln;         // this lane's query
  const __bf16* __restrict__ Qp = Qr + ((size_t)bh * Tc + q) * DHc;
  const __bf16* __restrict__ Kp = Kr + (size_t)bh * Tc * DHc;
  const __bf16* __restrict__ Vp = Vr + (size_t)bh * Tc * DHc;

  // LDS byte addresses (generic-pointer low 32 bits == wave-relative LDS address)
  const unsigned ldsK0 = (unsigned)(unsigned long long)(__bf16*)Ks[0];
  const unsigned ldsK1 = (unsigned)(unsigned long long)(__bf16*)Ks[1];
  const unsigned ldsV0 = (unsigned)(unsigned long long)(__bf16*)Vs[0];
  const unsigned ldsV1 = (unsigned)(unsigned long long)(__bf16*)Vs[1];

  // Q enters as B-fragments of Q^T (dh x queries): contiguous 32B per lane
  bf16x16 qf[4];
#pragma unroll
  for (int s4 = 0; s4 < 4; ++s4)
#pragma unroll
    for (int e = 0; e < 16; ++e) qf[s4][e] = Qp[s4 * 32 + lh * 16 + e];

  f32x8 o[8] = {};                              // O^T accumulators: 8 dh-chunks x 16 queries
  float m2 = -3.0e38f, l = 0.0f;
  const float sc2 = 1.44269504f * 0.08838834764831845f;  // log2(e) * DH^-0.5

  const int nkt = (qbase + QB) / KT;            // uniform across block (causal upper bound)

  // prologue: wave 0 kicks off tile 0 via the Tensor Data Mover
  if (wid == 0) {
    tdm_load_kv_tile(ldsK0, (unsigned long long)(const void*)Kp);
    tdm_load_kv_tile(ldsV0, (unsigned long long)(const void*)Vp);
  }

  for (int kt = 0; kt < nkt; ++kt) {
    const int kbase = kt * KT;
    const int cur = kt & 1;
    if (wid == 0) {
      if (kt + 1 < nkt) {   // double-buffer: issue next tile, then wait for current
        const size_t off = (size_t)(kbase + KT) * DHc;
        tdm_load_kv_tile(cur ? ldsK0 : ldsK1, (unsigned long long)(const void*)(Kp + off));
        tdm_load_kv_tile(cur ? ldsV0 : ldsV1, (unsigned long long)(const void*)(Vp + off));
        __builtin_amdgcn_s_wait_tensorcnt(2);   // TDM completes in-order per wave
      } else {
        __builtin_amdgcn_s_wait_tensorcnt(0);
      }
    }
    __syncthreads();
    const __bf16* __restrict__ Kt = Ks[cur];
    const __bf16* __restrict__ Vt = Vs[cur];

    // S^T = K (A) x Q^T (B): 2 key-groups x 4 dh-steps
    f32x8 sf[2] = {};
#pragma unroll
    for (int mt = 0; mt < 2; ++mt) {
#pragma unroll
      for (int s4 = 0; s4 < 4; ++s4) {
        bf16x16 a;
#pragma unroll
        for (int e = 0; e < 16; ++e) {
          int dh = s4 * 32 + 16 * lh + e;
          a[e] = Kt[(mt * 16 + ln) * LKV + dh];
        }
        sf[mt] = __builtin_amdgcn_wmma_f32_16x16x32_bf16(false, a, false, qf[s4], (short)0,
                                                         sf[mt], false, false);
      }
    }

    // causal mask + scale (log2 domain), running per-query max
    float tmax = -3.0e38f;
#pragma unroll
    for (int mt = 0; mt < 2; ++mt)
#pragma unroll
      for (int r = 0; r < 8; ++r) {
        int kg = kbase + mt * 16 + 8 * lh + r;
        float v = (kg <= q) ? sf[mt][r] * sc2 : -3.0e38f;
        sf[mt][r] = v;
        tmax = fmaxf(tmax, v);
      }
    tmax = fmaxf(tmax, __shfl_xor(tmax, 16, 32));
    float mnew = fmaxf(m2, tmax);
    float corr = exp2f(m2 - mnew);
    m2 = mnew;
    l = l * corr;
#pragma unroll
    for (int f = 0; f < 8; ++f)
#pragma unroll
      for (int r = 0; r < 8; ++r) o[f][r] = o[f][r] * corr;

    // P = exp2(s - m), row sum, re-pack P^T into a bf16 B-fragment via shuffles
    float p0[8], p1[8], psum = 0.0f;
#pragma unroll
    for (int r = 0; r < 8; ++r) {
      p0[r] = exp2f(sf[0][r] - m2);
      p1[r] = exp2f(sf[1][r] - m2);
      psum += p0[r] + p1[r];
    }
    psum += __shfl_xor(psum, 16, 32);
    l += psum;

    bf16x16 pb;
#pragma unroll
    for (int r = 0; r < 8; ++r) {
      float s0 = __shfl_xor(p0[r], 16, 32);
      float s1 = __shfl_xor(p1[r], 16, 32);
      pb[r]     = f2bf(lh ? s1 : p0[r]);   // B elem e<8 : key = 16*lh + e
      pb[r + 8] = f2bf(lh ? p1[r] : s0);   // B elem e>=8: key = 16*lh + 8 + (e-8)
    }

    // O^T += V^T (A) x P^T (B): 8 dh-chunks
#pragma unroll
    for (int f = 0; f < 8; ++f) {
      bf16x16 a;
#pragma unroll
      for (int e = 0; e < 16; ++e) {
        int key = 16 * lh + e;
        a[e] = Vt[key * LKV + f * 16 + ln];
      }
      o[f] = __builtin_amdgcn_wmma_f32_16x16x32_bf16(false, a, false, pb, (short)0,
                                                     o[f], false, false);
    }
    __syncthreads();   // protect buffer before it is overwritten by tile kt+2
  }

  // normalize and store heads output, bf16 [B,T,D] (16B packed stores)
  const float inv_l = 1.0f / l;
  const int b = bh >> 4, hh = bh & 15;
#pragma unroll
  for (int f = 0; f < 8; ++f) {
    unsigned w[4];
#pragma unroll
    for (int j = 0; j < 4; ++j) {
      unsigned short lo = __builtin_bit_cast(unsigned short, f2bf(o[f][2 * j] * inv_l));
      unsigned short hi = __builtin_bit_cast(unsigned short, f2bf(o[f][2 * j + 1] * inv_l));
      w[j] = (unsigned)lo | ((unsigned)hi << 16);
    }
    u32x4 vv = {w[0], w[1], w[2], w[3]};
    int dh0 = f * 16 + 8 * lh;
    __bf16* dp = Ob + ((size_t)(b * Tc + q) * Dc) + hh * 128 + dh0;
    *(u32x4*)dp = vv;
  }
}

// ---------------- kernel 4: output projection, fp32 result ----------------
__global__ __launch_bounds__(256) void mha_oproj(
    const __bf16* __restrict__ A, const __bf16* __restrict__ Wo, float* __restrict__ out) {
  __shared__ __align__(16) __bf16 Xs[BM * LDS_A];
  __shared__ __align__(16) __bf16 Ws[BN * LDS_B];

  const int tid = threadIdx.x, lane = tid & 31, wid = tid >> 5;
  const int lh = lane >> 4, ln = lane & 15;
  const int waveM = wid >> 1, waveN = wid & 1;
  const int mBase = blockIdx.y * BM;
  const int nBase = blockIdx.x * BN;

  f32x8 acc[4] = {};
  for (int k0 = 0; k0 < Dc; k0 += BK) {
    {
      int row = tid >> 2, c = (tid & 3) * 8;
      *(u32x4*)(Xs + row * LDS_A + c) = *(const u32x4*)(A + (size_t)(mBase + row) * Dc + k0 + c);
    }
#pragma unroll
    for (int i = 0; i < 2; ++i) {
      int c2 = tid + i * 256;
      int row = c2 >> 2, c = (c2 & 3) * 8;
      *(u32x4*)(Ws + row * LDS_B + c) = *(const u32x4*)(Wo + (size_t)(nBase + row) * Dc + k0 + c);
    }
    __syncthreads();
    if (k0 + BK < Dc) {
      __builtin_prefetch(A + (size_t)(mBase + (tid >> 2)) * Dc + k0 + BK, 0, 1);
      __builtin_prefetch(Wo + (size_t)(nBase + (tid >> 1)) * Dc + k0 + BK, 0, 1);
    }
    bf16x16 a;
#pragma unroll
    for (int e = 0; e < 16; ++e) {
      int kk = 16 * lh + e;
      a[e] = Xs[(waveM * 16 + ln) * LDS_A + kk];
    }
#pragma unroll
    for (int nf = 0; nf < 4; ++nf) {
      bf16x16 bf;
#pragma unroll
      for (int e = 0; e < 16; ++e) bf[e] = Ws[(waveN * 64 + nf * 16 + ln) * LDS_B + 16 * lh + e];
      acc[nf] = __builtin_amdgcn_wmma_f32_16x16x32_bf16(false, a, false, bf, (short)0,
                                                        acc[nf], false, false);
    }
    __syncthreads();
  }
#pragma unroll
  for (int nf = 0; nf < 4; ++nf) {
    int n = nBase + waveN * 64 + nf * 16 + ln;
#pragma unroll
    for (int r = 0; r < 8; ++r) {
      int mtok = mBase + waveM * 16 + 8 * lh + r;
      out[(size_t)mtok * Dc + n] = acc[nf][r];
    }
  }
}

// ---------------- host launcher ----------------
extern "C" void kernel_launch(void* const* d_in, const int* in_sizes, int n_in,
                              void* d_out, int out_size, void* d_ws, size_t ws_size,
                              hipStream_t stream) {
  (void)in_sizes; (void)n_in; (void)out_size; (void)ws_size;
  const float* X  = (const float*)d_in[0];
  const float* Wq = (const float*)d_in[1];
  const float* Wk = (const float*)d_in[2];
  const float* Wv = (const float*)d_in[3];
  const float* Wo = (const float*)d_in[4];
  float* out = (float*)d_out;

  char* ws = (char*)d_ws;
  const size_t MB = 1ull << 20;
  __bf16* Xb  = (__bf16*)(ws + 0 * MB);    // 16 MB (reused as Ob after QKV)
  __bf16* Wqb = (__bf16*)(ws + 16 * MB);   //  8 MB each
  __bf16* Wkb = (__bf16*)(ws + 24 * MB);
  __bf16* Wvb = (__bf16*)(ws + 32 * MB);
  __bf16* Wob = (__bf16*)(ws + 40 * MB);
  __bf16* Qr  = (__bf16*)(ws + 48 * MB);   // 16 MB each, [B,H,T,DH]
  __bf16* Kr  = (__bf16*)(ws + 64 * MB);
  __bf16* Vr  = (__bf16*)(ws + 80 * MB);
  __bf16* Ob  = Xb;                        // reuse: X no longer needed after QKV

  const int nX = Bc * Tc * Dc;             // 8388608
  const int nW = Dc * Dc;                  // 4194304
  mha_cvt_bf16<<<nX / 1024, 256, 0, stream>>>(X,  Xb,  nX);
  mha_cvt_bf16<<<nW / 1024, 256, 0, stream>>>(Wq, Wqb, nW);
  mha_cvt_bf16<<<nW / 1024, 256, 0, stream>>>(Wk, Wkb, nW);
  mha_cvt_bf16<<<nW / 1024, 256, 0, stream>>>(Wv, Wvb, nW);
  mha_cvt_bf16<<<nW / 1024, 256, 0, stream>>>(Wo, Wob, nW);

  dim3 gProj(Dc / BN, (Bc * Tc) / BM, 3);  // (16, 64, 3)
  mha_qkv_rope<<<gProj, 256, 0, stream>>>(Xb, Wqb, Wkb, Wvb, Qr, Kr, Vr);

  dim3 gAttn(Tc / QB, Bc * Hc);            // (16, 32)
  mha_attn<<<gAttn, 256, 0, stream>>>(Qr, Kr, Vr, Ob);

  dim3 gO(Dc / BN, (Bc * Tc) / BM);        // (16, 64)
  mha_oproj<<<gO, 256, 0, stream>>>(Ob, Wob, out);
}
